// AttentionModule_86552180949475
// MI455X (gfx1250) — matrix operand order repaired and checked by reading, not compiled, not run on hardware
//
#include <hip/hip_runtime.h>
#include <hip/hip_bf16.h>

typedef __attribute__((ext_vector_type(16))) __bf16 v16bf;
typedef __attribute__((ext_vector_type(8)))  __bf16 v8bf;
typedef __attribute__((ext_vector_type(8)))  float  v8f;

#define N_TOK   4096
#define C_CH    256
#define D_QK    32
#define PSTRIDE 264   // 256 + 8 pad -> lane stride 132 dwords, conflict-free b128

__device__ __forceinline__ v16bf cat8(v8bf lo, v8bf hi) {
    union { v16bf v; v8bf h[2]; } u;
    u.h[0] = lo; u.h[1] = hi;
    return u.v;
}

// ---------------------------------------------------------------------------
// Projection kernel: Q = Wq*x + bq (token-major bf16 [B][N][32])
//                    K = Wk*x + bk (token-major bf16 [B][N][32])
//                    V = Wv*x + bv (feature-major bf16 [B][256][N])
// chunk 0 -> Q, chunk 1 -> K, chunks 2..9 -> V feature blocks of 32.
// ---------------------------------------------------------------------------
__global__ __launch_bounds__(256)
void proj_kernel(const float* __restrict__ x,
                 const float* __restrict__ Wq, const float* __restrict__ bq,
                 const float* __restrict__ Wk, const float* __restrict__ bk,
                 const float* __restrict__ Wv, const float* __restrict__ bv,
                 __bf16* __restrict__ Qst, __bf16* __restrict__ Kst,
                 __bf16* __restrict__ Vt) {
    __shared__ float smem[32 * 256];   // 32KB weight tile
    const int tid   = threadIdx.x;
    const int chunk = blockIdx.x % 10;
    const int nb    = (blockIdx.x / 10) % (N_TOK / 256);
    const int b     = blockIdx.x / (10 * (N_TOK / 256));
    const int n     = nb * 256 + tid;

    const float* W;
    const float* bias;
    int o0 = 0;
    if (chunk == 0)      { W = Wq; bias = bq; }
    else if (chunk == 1) { W = Wk; bias = bk; }
    else                 { o0 = (chunk - 2) * 32; W = Wv + (size_t)o0 * 256; bias = bv + o0; }

    for (int i = tid; i < 32 * 256; i += 256) smem[i] = W[i];
    __syncthreads();

    float acc[32];
#pragma unroll
    for (int o = 0; o < 32; ++o) acc[o] = 0.f;

    const float* xp = x + (size_t)b * C_CH * N_TOK + n;
    for (int c = 0; c < C_CH; ++c) {
        float xv = xp[(size_t)c * N_TOK];          // coalesced across threads
#pragma unroll
        for (int o = 0; o < 32; ++o) acc[o] += smem[o * 256 + c] * xv;  // LDS broadcast
    }
#pragma unroll
    for (int o = 0; o < 32; ++o) acc[o] += bias[o];

    if (chunk < 2) {
        __bf16* dst = (chunk == 0 ? Qst : Kst) + ((size_t)(b * N_TOK + n)) * D_QK;
#pragma unroll
        for (int o = 0; o < 32; ++o) dst[o] = (__bf16)acc[o];   // 64B contiguous/thread
    } else {
#pragma unroll
        for (int o = 0; o < 32; ++o)
            Vt[((size_t)(b * C_CH + o0 + o)) * N_TOK + n] = (__bf16)acc[o]; // coalesced in n
    }
}

// ---------------------------------------------------------------------------
// Flash-attention kernel (no max-subtraction: logits bounded ~|50| -> f32 safe)
// One workgroup (8 waves) = one 32-query tile (halves K/V L2 traffic vs 16).
// Per 256-key block:
//   wave w computes exp(Q*K^T) for its 32 keys x 32 queries -> P in LDS (4 WMMA)
//   wave w accumulates O[32q, 32w..32w+31] += P(32x256)*V(256x32) (32 WMMA),
//   with each V B-fragment reused by both query-row tiles.
// ---------------------------------------------------------------------------
__global__ __launch_bounds__(256)
void attn_kernel(const __bf16* __restrict__ Qst, const __bf16* __restrict__ Kst,
                 const __bf16* __restrict__ Vt, float* __restrict__ out) {
    __shared__ __bf16 Plds[32 * PSTRIDE];
    __shared__ float  rowsum[32];

    const int tid  = threadIdx.x;
    const int wave = tid >> 5;
    const int lane = tid & 31;
    const int half = lane >> 4;
    const int l16  = lane & 15;
    const int b    = blockIdx.x >> 7;            // N_TOK/32 = 128 tiles per batch
    const int q0   = (blockIdx.x & 127) << 5;

    if (tid < 32) rowsum[tid] = 0.f;

    // Q A-fragments: two 16-row tiles (layout per ISA 16-bit A-matrix table)
    v16bf aQ[2];
#pragma unroll
    for (int qt = 0; qt < 2; ++qt) {
        const __bf16* qrow = Qst + ((size_t)(b * N_TOK) + q0 + (qt << 4) + l16) * D_QK;
        aQ[qt] = cat8(*(const v8bf*)(qrow + (half ? 8 : 0)),
                      *(const v8bf*)(qrow + (half ? 24 : 16)));
    }

    const v8f zero = {};
    v8f oacc[2][2];
#pragma unroll
    for (int qt = 0; qt < 2; ++qt) { oacc[qt][0] = zero; oacc[qt][1] = zero; }
    const int c0 = wave << 5;                    // this wave's 32 output channels

    for (int kb = 0; kb < N_TOK; kb += 256) {
        float ex[2][2][8];                       // [qt][key tile][j]
        float psum[2][8];
#pragma unroll
        for (int qt = 0; qt < 2; ++qt)
#pragma unroll
            for (int j = 0; j < 8; ++j) psum[qt][j] = 0.f;

        // S = Q (32x32) x K^T (32x32) for this wave's keys -> exp
#pragma unroll
        for (int t = 0; t < 2; ++t) {
            const size_t keyrow = (size_t)(b * N_TOK) + kb + (wave << 5) + (t << 4) + l16;
            const __bf16* krow  = Kst + keyrow * D_QK + (half << 3);
            const v16bf bK = cat8(*(const v8bf*)krow, *(const v8bf*)(krow + 16));
#pragma unroll
            for (int qt = 0; qt < 2; ++qt) {
                v8f s = __builtin_amdgcn_wmma_f32_16x16x32_bf16(
                            false, aQ[qt], false, bK, (short)0, zero, false, false);
#pragma unroll
                for (int j = 0; j < 8; ++j) {
                    float e = __expf(s[j]);
                    ex[qt][t][j] = e;
                    psum[qt][j] += e;
                }
            }
        }

        // reduce partial rowsums over 16 key-columns (lanes within a half)
#pragma unroll
        for (int qt = 0; qt < 2; ++qt)
#pragma unroll
            for (int j = 0; j < 8; ++j) {
                float p = psum[qt][j];
                p += __shfl_xor(p, 1);
                p += __shfl_xor(p, 2);
                p += __shfl_xor(p, 4);
                p += __shfl_xor(p, 8);
                psum[qt][j] = p;
            }

        __syncthreads();   // previous iteration's P reads complete

#pragma unroll
        for (int qt = 0; qt < 2; ++qt)
#pragma unroll
            for (int t = 0; t < 2; ++t)
#pragma unroll
                for (int j = 0; j < 8; ++j)
                    Plds[((qt << 4) + j + (half << 3)) * PSTRIDE +
                         (wave << 5) + (t << 4) + l16] = (__bf16)ex[qt][t][j];
        if (l16 == 0) {
#pragma unroll
            for (int qt = 0; qt < 2; ++qt)
#pragma unroll
                for (int j = 0; j < 8; ++j)
                    atomicAdd(&rowsum[(qt << 4) + j + (half << 3)], psum[qt][j]);
        }

        __syncthreads();   // P tile ready

        // O += P(32x256) * V(256x32): each bV reused by both query tiles
#pragma unroll
        for (int ct = 0; ct < 2; ++ct) {
            const __bf16* vrow = Vt + ((size_t)(b * C_CH) + c0 + (ct << 4) + l16) * N_TOK
                                    + kb + (half << 3);
#pragma unroll
            for (int kblk = 0; kblk < 8; ++kblk) {
                const v16bf bV = cat8(*(const v8bf*)(vrow + (kblk << 5)),
                                      *(const v8bf*)(vrow + (kblk << 5) + 16));
#pragma unroll
                for (int qt = 0; qt < 2; ++qt) {
                    const __bf16* pr = &Plds[((qt << 4) + l16) * PSTRIDE + (kblk << 5)];
                    const v16bf aP = cat8(*(const v8bf*)(pr + (half ? 8 : 0)),
                                          *(const v8bf*)(pr + (half ? 24 : 16)));
                    oacc[qt][ct] = __builtin_amdgcn_wmma_f32_16x16x32_bf16(
                                       false, aP, false, bV, (short)0, oacc[qt][ct],
                                       false, false);
                }
            }
        }
    }

    __syncthreads();
    float inv[2][8];
#pragma unroll
    for (int qt = 0; qt < 2; ++qt)
#pragma unroll
        for (int j = 0; j < 8; ++j)
            inv[qt][j] = 1.0f / rowsum[(qt << 4) + j + (half << 3)];

    // out[b][col][q0 + 16*qt + m], m = j + 8*half -> 32B contiguous per lane
#pragma unroll
    for (int qt = 0; qt < 2; ++qt)
#pragma unroll
        for (int ct = 0; ct < 2; ++ct) {
            const int col = c0 + (ct << 4) + l16;
            float* op = out + ((size_t)(b * C_CH) + col) * N_TOK + q0 + (qt << 4) + (half << 3);
#pragma unroll
            for (int j = 0; j < 8; ++j) op[j] = oacc[qt][ct][j] * inv[qt][j];
        }
}

// ---------------------------------------------------------------------------
extern "C" void kernel_launch(void* const* d_in, const int* in_sizes, int n_in,
                              void* d_out, int out_size, void* d_ws, size_t ws_size,
                              hipStream_t stream) {
    (void)in_sizes; (void)n_in; (void)out_size; (void)ws_size;
    const float* x  = (const float*)d_in[0];
    const float* Wq = (const float*)d_in[1];
    const float* bq = (const float*)d_in[2];
    const float* Wk = (const float*)d_in[3];
    const float* bk = (const float*)d_in[4];
    const float* Wv = (const float*)d_in[5];
    const float* bv = (const float*)d_in[6];

    const int B = 4;
    __bf16* Qst = (__bf16*)d_ws;                                   // B*N*32
    __bf16* Kst = Qst + (size_t)B * N_TOK * D_QK;                  // B*N*32
    __bf16* Vt  = Kst + (size_t)B * N_TOK * D_QK;                  // B*256*N

    proj_kernel<<<B * (N_TOK / 256) * 10, 256, 0, stream>>>(
        x, Wq, bq, Wk, bk, Wv, bv, Qst, Kst, Vt);
    attn_kernel<<<B * (N_TOK / 32), 256, 0, stream>>>(
        Qst, Kst, Vt, (float*)d_out);
}
